// MixtureOfExperts_30906584662358
// MI455X (gfx1250) — matrix operand order repaired and checked by reading, not compiled, not run on hardware
//
#include <hip/hip_runtime.h>
#include <hip/hip_bf16.h>

#define D_MODEL 1024
#define D_FF    4096
#define NUM_EXP 8
#define TOKENS  4096
#define TOPK    2

typedef __bf16 bf16_t;
typedef __attribute__((ext_vector_type(16))) __bf16 v16bf;
typedef __attribute__((ext_vector_type(8)))  __bf16 v8bf;
typedef __attribute__((ext_vector_type(8)))  float  v8f;

#define WMMA_BF16(a, b, c) \
    __builtin_amdgcn_wmma_f32_16x16x32_bf16(false, (a), false, (b), (short)0, (c), false, false)

// ---------------- fp32 -> bf16 (row-major, for activations) ----------------
__global__ void moe_cvt_bf16(const float* __restrict__ src, bf16_t* __restrict__ dst, long n) {
    long i = ((long)blockIdx.x * blockDim.x + threadIdx.x) * 8;
    if (i >= n) return;
    float4 a = *(const float4*)(src + i);
    float4 b = *(const float4*)(src + i + 4);
    v8bf o;
    o[0] = (bf16_t)a.x; o[1] = (bf16_t)a.y; o[2] = (bf16_t)a.z; o[3] = (bf16_t)a.w;
    o[4] = (bf16_t)b.x; o[5] = (bf16_t)b.y; o[6] = (bf16_t)b.z; o[7] = (bf16_t)b.w;
    *(v8bf*)(dst + i) = o;
}

// ------- fp32 [R][C] -> bf16 [C][R] transpose-convert (32x32 LDS tiles) -------
__global__ void moe_cvt_t(const float* __restrict__ src, bf16_t* __restrict__ dst,
                          int R, int C) {
    __shared__ float tile[32][33];
    const long esz = (long)R * C;
    src += (long)blockIdx.z * esz;
    dst += (long)blockIdx.z * esz;
    const int c0 = blockIdx.x * 32, r0 = blockIdx.y * 32;
    const int tx = threadIdx.x, ty = threadIdx.y;      // 32 x 8
#pragma unroll
    for (int i = 0; i < 32; i += 8)
        tile[ty + i][tx] = src[(long)(r0 + ty + i) * C + c0 + tx];
    __syncthreads();
#pragma unroll
    for (int i = 0; i < 32; i += 8)
        dst[(long)(c0 + ty + i) * R + r0 + tx] = (bf16_t)tile[tx][ty + i];
}

// ---------------- gating: one wave per token, top-2 of logits ----------------
// Also records (expert, slot) pairs per token so the combine pass needs no atomics.
__global__ void moe_gate(const float* __restrict__ x, const float* __restrict__ Wg,
                         const float* __restrict__ bg,
                         int* __restrict__ counts, int* __restrict__ lists,
                         int* __restrict__ info) {
    int wv   = (int)((blockIdx.x * blockDim.x + threadIdx.x) >> 5);
    int lane = threadIdx.x & 31;
    if (wv >= TOKENS) return;
    float acc[NUM_EXP];
#pragma unroll
    for (int e = 0; e < NUM_EXP; ++e) acc[e] = 0.f;
    const float* xr = x + (long)wv * D_MODEL;
    for (int d = lane; d < D_MODEL; d += 32) {
        float xv = xr[d];
        const float* wr = Wg + (long)d * NUM_EXP;
#pragma unroll
        for (int e = 0; e < NUM_EXP; ++e) acc[e] += xv * wr[e];
    }
#pragma unroll
    for (int e = 0; e < NUM_EXP; ++e) {
#pragma unroll
        for (int off = 16; off > 0; off >>= 1)
            acc[e] += __shfl_xor(acc[e], off, 32);
        acc[e] += bg[e];
    }
    if (lane == 0) {
        int i1 = 0; float v1 = acc[0];
#pragma unroll
        for (int e = 1; e < NUM_EXP; ++e) if (acc[e] > v1) { v1 = acc[e]; i1 = e; }
        int i2 = -1; float v2 = -3.4e38f;
#pragma unroll
        for (int e = 0; e < NUM_EXP; ++e) if (e != i1 && acc[e] > v2) { v2 = acc[e]; i2 = e; }
        int p1 = atomicAdd(&counts[i1], 1);
        lists[(long)i1 * TOKENS + p1] = wv;
        int p2 = atomicAdd(&counts[i2], 1);
        lists[(long)i2 * TOKENS + p2] = wv;
        int4 inf; inf.x = i1; inf.y = p1; inf.z = i2; inf.w = p2;
        *(int4*)(info + 4 * wv) = inf;
    }
}

__global__ void moe_scan(const int* __restrict__ counts, int* __restrict__ offsets) {
    if (threadIdx.x == 0) {
        int s = 0;
        for (int e = 0; e < NUM_EXP; ++e) { offsets[e] = s; s += counts[e]; }
    }
}

// ---------------- GEMM1: H = relu(gather(Xb) @ W1 + b1), wave tile 64x64 ----------------
// A frag (16-bit 16x32): lane row = lane&15; 16 consecutive K at k0+h*16 (two 16B loads).
// B frag from [N][K] layout: lane col = lane&15; 16 consecutive K at k0+(lane>>4)*16 (32B load).
// C frag: n = lane&15, m = r + 8*(lane>>4).
__global__ void __launch_bounds__(64)
moe_gemm1(const bf16_t* __restrict__ Xb, const bf16_t* __restrict__ W1t,
          const float* __restrict__ b1,
          const int* __restrict__ counts, const int* __restrict__ offsets,
          const int* __restrict__ lists, bf16_t* __restrict__ H) {
    const int e  = blockIdx.z;
    const int ne = counts[e];
    const int wave = threadIdx.x >> 5;
    const int m0 = blockIdx.y * 128 + wave * 64;
    if (m0 >= ne) return;
    const int n0 = blockIdx.x * 64;
    const int lane = threadIdx.x & 31;
    const int half = lane >> 4;
    const int rsel = lane & 15;

    const bf16_t* aptr[4];
#pragma unroll
    for (int ms = 0; ms < 4; ++ms) {
        int r = m0 + ms * 16 + rsel; if (r >= ne) r = ne - 1;
        aptr[ms] = Xb + (long)lists[(long)e * TOKENS + r] * D_MODEL + half * 16;
    }
    const bf16_t* bbase = W1t + (long)e * D_MODEL * D_FF;  // [D_FF][D_MODEL]
    const bf16_t* bptr[4];
#pragma unroll
    for (int ns = 0; ns < 4; ++ns)
        bptr[ns] = bbase + (long)(n0 + ns * 16 + rsel) * D_MODEL + half * 16;

    v8f acc[4][4];
#pragma unroll
    for (int ms = 0; ms < 4; ++ms)
#pragma unroll
        for (int ns = 0; ns < 4; ++ns) acc[ms][ns] = (v8f){};

    for (int k0 = 0; k0 < D_MODEL; k0 += 32) {
        v16bf b[4];
#pragma unroll
        for (int ns = 0; ns < 4; ++ns)
            b[ns] = *(const v16bf*)(bptr[ns] + k0);
#pragma unroll
        for (int ms = 0; ms < 4; ++ms) {
            v8bf lo = *(const v8bf*)(aptr[ms] + k0);
            v8bf hi = *(const v8bf*)(aptr[ms] + k0 + 8);
            v16bf a = __builtin_shufflevector(lo, hi, 0,1,2,3,4,5,6,7,8,9,10,11,12,13,14,15);
#pragma unroll
            for (int ns = 0; ns < 4; ++ns)
                acc[ms][ns] = WMMA_BF16(a, b[ns], acc[ms][ns]);
        }
    }

    const int base = offsets[e];
#pragma unroll
    for (int ms = 0; ms < 4; ++ms) {
#pragma unroll
        for (int rr = 0; rr < 8; ++rr) {
            int row = m0 + ms * 16 + rr + 8 * half;
            if (row < ne) {
                bf16_t* hrow = H + (long)(base + row) * D_FF + n0;
#pragma unroll
                for (int ns = 0; ns < 4; ++ns) {
                    float v = acc[ms][ns][rr] + b1[(long)e * D_FF + n0 + ns * 16 + rsel];
                    v = v > 0.f ? v : 0.f;
                    hrow[ns * 16 + rsel] = (bf16_t)v;
                }
            }
        }
    }
}

// ---------------- GEMM2: Y[row] = H_e @ W2 (no bias, plain stores), wave tile 64x64 ----------------
__global__ void __launch_bounds__(64)
moe_gemm2(const bf16_t* __restrict__ H, const bf16_t* __restrict__ W2t,
          const int* __restrict__ counts, const int* __restrict__ offsets,
          float* __restrict__ Y) {
    const int e  = blockIdx.z;
    const int ne = counts[e];
    const int wave = threadIdx.x >> 5;
    const int m0 = blockIdx.y * 128 + wave * 64;
    if (m0 >= ne) return;
    const int n0 = blockIdx.x * 64;
    const int lane = threadIdx.x & 31;
    const int half = lane >> 4;
    const int rsel = lane & 15;
    const int base = offsets[e];

    const bf16_t* aptr[4];
#pragma unroll
    for (int ms = 0; ms < 4; ++ms) {
        int r = m0 + ms * 16 + rsel; if (r >= ne) r = ne - 1;
        aptr[ms] = H + (long)(base + r) * D_FF + half * 16;
    }
    const bf16_t* bbase = W2t + (long)e * D_FF * D_MODEL;  // [D_MODEL][D_FF]
    const bf16_t* bptr[4];
#pragma unroll
    for (int ns = 0; ns < 4; ++ns)
        bptr[ns] = bbase + (long)(n0 + ns * 16 + rsel) * D_FF + half * 16;

    v8f acc[4][4];
#pragma unroll
    for (int ms = 0; ms < 4; ++ms)
#pragma unroll
        for (int ns = 0; ns < 4; ++ns) acc[ms][ns] = (v8f){};

    for (int k0 = 0; k0 < D_FF; k0 += 32) {
        v16bf b[4];
#pragma unroll
        for (int ns = 0; ns < 4; ++ns)
            b[ns] = *(const v16bf*)(bptr[ns] + k0);
#pragma unroll
        for (int ms = 0; ms < 4; ++ms) {
            v8bf lo = *(const v8bf*)(aptr[ms] + k0);
            v8bf hi = *(const v8bf*)(aptr[ms] + k0 + 8);
            v16bf a = __builtin_shufflevector(lo, hi, 0,1,2,3,4,5,6,7,8,9,10,11,12,13,14,15);
#pragma unroll
            for (int ns = 0; ns < 4; ++ns)
                acc[ms][ns] = WMMA_BF16(a, b[ns], acc[ms][ns]);
        }
    }

#pragma unroll
    for (int ms = 0; ms < 4; ++ms) {
#pragma unroll
        for (int rr = 0; rr < 8; ++rr) {
            int row = m0 + ms * 16 + rr + 8 * half;
            if (row < ne) {
                float* yrow = Y + (long)(base + row) * D_MODEL + n0;
#pragma unroll
                for (int ns = 0; ns < 4; ++ns)
                    yrow[ns * 16 + rsel] = acc[ms][ns][rr];
            }
        }
    }
}

// ---------------- combine: out[t] = Y[r1] + Y[r2] + b2[e1] + b2[e2] ----------------
__global__ void moe_combine(const float* __restrict__ Y, const int* __restrict__ info,
                            const float* __restrict__ b2, const int* __restrict__ offsets,
                            float* __restrict__ out) {
    long gid = (long)blockIdx.x * blockDim.x + threadIdx.x;
    int t  = (int)(gid / (D_MODEL / 4));
    int c4 = (int)(gid % (D_MODEL / 4)) * 4;
    if (t >= TOKENS) return;
    const int4 inf = *(const int4*)(info + 4 * t);
    long r1 = (long)offsets[inf.x] + inf.y;
    long r2 = (long)offsets[inf.z] + inf.w;
    float4 y1 = *(const float4*)(Y + r1 * D_MODEL + c4);
    float4 y2 = *(const float4*)(Y + r2 * D_MODEL + c4);
    float4 c1 = *(const float4*)(b2 + (long)inf.x * D_MODEL + c4);
    float4 c2 = *(const float4*)(b2 + (long)inf.z * D_MODEL + c4);
    float4 o;
    o.x = y1.x + y2.x + c1.x + c2.x;
    o.y = y1.y + y2.y + c1.y + c2.y;
    o.z = y1.z + y2.z + c1.z + c2.z;
    o.w = y1.w + y2.w + c1.w + c2.w;
    *(float4*)(out + (long)t * D_MODEL + c4) = o;
}

// ---------------- host-side orchestration ----------------
extern "C" void kernel_launch(void* const* d_in, const int* in_sizes, int n_in,
                              void* d_out, int out_size, void* d_ws, size_t ws_size,
                              hipStream_t stream) {
    (void)in_sizes; (void)n_in; (void)ws_size; (void)out_size;
    const float* x  = (const float*)d_in[0];
    const float* Wg = (const float*)d_in[1];
    const float* bg = (const float*)d_in[2];
    const float* W1 = (const float*)d_in[3];
    const float* b1 = (const float*)d_in[4];
    const float* W2 = (const float*)d_in[5];
    const float* b2 = (const float*)d_in[6];
    float* out = (float*)d_out;

    char* ws = (char*)d_ws;
    const long XB_N = (long)TOKENS * D_MODEL;
    const long W_N  = (long)NUM_EXP * D_MODEL * D_FF;
    const long H_N  = (long)TOKENS * TOPK * D_FF;
    const long Y_N  = (long)TOKENS * TOPK * D_MODEL;
    bf16_t* Xb  = (bf16_t*)ws;  ws += XB_N * sizeof(bf16_t);
    bf16_t* W1t = (bf16_t*)ws;  ws += W_N  * sizeof(bf16_t);   // [E][D_FF][D_MODEL]
    bf16_t* W2t = (bf16_t*)ws;  ws += W_N  * sizeof(bf16_t);   // [E][D_MODEL][D_FF]
    bf16_t* H   = (bf16_t*)ws;  ws += H_N  * sizeof(bf16_t);
    float*  Y   = (float*)ws;   ws += Y_N  * sizeof(float);
    int* counts  = (int*)ws;    ws += 256;
    int* offsets = (int*)ws;    ws += 256;
    int* lists   = (int*)ws;    ws += (long)NUM_EXP * TOKENS * sizeof(int);
    int* info    = (int*)ws;    ws += (long)TOKENS * 4 * sizeof(int);

    hipMemsetAsync(counts, 0, NUM_EXP * sizeof(int), stream);

    // activations: plain convert (K-contiguous already)
    moe_cvt_bf16<<<(unsigned)((XB_N / 8 + 255) / 256), 256, 0, stream>>>(x, Xb, XB_N);
    // weights: transpose-convert into [N][K] so B-frags are contiguous 32B loads
    {
        dim3 b(32, 8);
        dim3 g1(D_FF / 32, D_MODEL / 32, NUM_EXP);   // W1 [D][F] -> [F][D]
        moe_cvt_t<<<g1, b, 0, stream>>>(W1, W1t, D_MODEL, D_FF);
        dim3 g2(D_MODEL / 32, D_FF / 32, NUM_EXP);   // W2 [F][D] -> [D][F]
        moe_cvt_t<<<g2, b, 0, stream>>>(W2, W2t, D_FF, D_MODEL);
    }

    moe_gate<<<(TOKENS * 32) / 256, 256, 0, stream>>>(x, Wg, bg, counts, lists, info);
    moe_scan<<<1, 32, 0, stream>>>(counts, offsets);

    dim3 g1(D_FF / 64, TOKENS / 128, NUM_EXP);
    moe_gemm1<<<g1, 64, 0, stream>>>(Xb, W1t, b1, counts, offsets, lists, H);
    dim3 g2(D_MODEL / 64, TOKENS / 128, NUM_EXP);
    moe_gemm2<<<g2, 64, 0, stream>>>(H, W2t, counts, offsets, Y);

    long cthreads = (long)TOKENS * (D_MODEL / 4);
    moe_combine<<<(unsigned)((cthreads + 255) / 256), 256, 0, stream>>>(Y, info, b2, offsets, out);
}
